// BatchChildSumTreeLSTM_24086176596339
// MI455X (gfx1250) — matrix operand order repaired and checked by reading, not compiled, not run on hardware
//
#include <hip/hip_runtime.h>
#include <hip/hip_bf16.h>
#include <math.h>

// Problem dims (fixed by the reference)
#define Bv   128
#define Sv   64
#define Ev   300
#define EPv  320           // E padded to multiple of 32 for WMMA K
#define Hv   512
#define Lv   5
#define Mx   (Bv * Sv)     // 8192 rows for the x-side GEMM

typedef __attribute__((ext_vector_type(16))) __bf16 v16bf;
typedef __attribute__((ext_vector_type(8)))  __bf16 v8bf;
typedef __attribute__((ext_vector_type(8)))  float  v8f;

__device__ __forceinline__ __bf16 f2bf(float f) {
  unsigned u = __builtin_bit_cast(unsigned, f);
  u += 0x7FFFu + ((u >> 16) & 1u);          // round-to-nearest-even
  unsigned short h = (unsigned short)(u >> 16);
  return __builtin_bit_cast(__bf16, h);
}

__device__ __forceinline__ float sigmoidf_(float x) {
  return 1.0f / (1.0f + __expf(-x));
}

// -------------------------------------------------------------------------
// CDNA5 16-bit WMMA fragment loaders (ISA 7.12.2 layouts):
//   A (16x32, row-major source): lane&15 = row M; elems 0..7 = K{0..7} (lo
//   lanes) / K{8..15} (hi lanes); elems 8..15 = K{16..23}/{24..31}.
//   B (32x16) from weights in ORIGINAL [out=n, in=k] row-major (B = W^T):
//   lane&15 = column N; same K order as A: elems 0..7 = K{0..7}/K{8..15},
//   elems 8..15 = K{16..23}/{24..31}.
// -------------------------------------------------------------------------
__device__ __forceinline__ v16bf ld_afrag(const __bf16* p) {
  v8bf lo = *(const v8bf*)p;
  v8bf hi = *(const v8bf*)(p + 16);
  return __builtin_shufflevector(lo, hi, 0,1,2,3,4,5,6,7,8,9,10,11,12,13,14,15);
}
__device__ __forceinline__ v16bf ld_bfrag(const __bf16* p) {
  v8bf lo = *(const v8bf*)p;
  v8bf hi = *(const v8bf*)(p + 16);
  return __builtin_shufflevector(lo, hi, 0,1,2,3,4,5,6,7,8,9,10,11,12,13,14,15);
}

#define WMMA_BF16(ACC, AV, BV)                                            \
  (ACC) = __builtin_amdgcn_wmma_f32_16x16x32_bf16(false, (AV), false,     \
                                                  (BV), (short)0, (ACC),  \
                                                  false, false)

// One 32x32 output block per wave (2x2 tiles, 4 accumulators).
// 8 b128 loads feed 4 WMMAs per k-step -> 2x data reuse vs 1 tile/wave.
__device__ __forceinline__ void wmma_block32(
    const __bf16* __restrict__ A, int lda,
    const __bf16* __restrict__ W, int ldw,
    int m0, int n0, int K, int lane,
    v8f& acc00, v8f& acc01, v8f& acc10, v8f& acc11) {
  const int rlo   = lane & 15;
  const int akoff = (lane >> 4) << 3;   // 0 or 8
  const int bkoff = (lane >> 4) << 3;   // 0 or 8
  const __bf16* pa0 = A + (size_t)(m0 + rlo) * lda + akoff;
  const __bf16* pa1 = pa0 + (size_t)16 * lda;
  const __bf16* pb0 = W + (size_t)(n0 + rlo) * ldw + bkoff;
  const __bf16* pb1 = pb0 + (size_t)16 * ldw;
#pragma unroll 2
  for (int k0 = 0; k0 < K; k0 += 32) {
    v16bf a0 = ld_afrag(pa0 + k0);
    v16bf a1 = ld_afrag(pa1 + k0);
    v16bf b0 = ld_bfrag(pb0 + k0);
    v16bf b1 = ld_bfrag(pb1 + k0);
    WMMA_BF16(acc00, a0, b0);
    WMMA_BF16(acc01, a0, b1);
    WMMA_BF16(acc10, a1, b0);
    WMMA_BF16(acc11, a1, b1);
  }
}

// ---------------- staging kernels ----------------------------------------

// XEb[(b*S+s), k] = bf16(embed[x[b,s], k]) for k<E else 0   (K-padded)
__global__ __launch_bounds__(256)
void embed_gather_kernel(const float* __restrict__ embed, const int* __restrict__ x,
                         __bf16* __restrict__ XEb) {
  int idx = blockIdx.x * 256 + threadIdx.x;           // Mx*EPv total, exact
  int row = idx / EPv;
  int k   = idx - row * EPv;
  int word = x[row];
  float v = (k < Ev) ? embed[(size_t)word * Ev + k] : 0.0f;
  XEb[idx] = f2bf(v);
}

// WXb[g][n][k] = bf16(W_gx[n][k]) for k<E else 0
__global__ __launch_bounds__(256)
void convert_wx_kernel(const float* __restrict__ Wix, const float* __restrict__ Wfx,
                       const float* __restrict__ Wux, __bf16* __restrict__ WXb) {
  int idx = blockIdx.x * 256 + threadIdx.x;           // 3*Hv*EPv total, exact
  int gate = idx / (Hv * EPv);
  int rem  = idx - gate * (Hv * EPv);
  int n = rem / EPv;
  int k = rem - n * EPv;
  const float* W = (gate == 0) ? Wix : (gate == 1) ? Wfx : Wux;
  float v = (k < Ev) ? W[(size_t)n * Ev + k] : 0.0f;
  WXb[idx] = f2bf(v);
}

// WHb[g][n][k] = bf16(W_gh[n][k])
__global__ __launch_bounds__(256)
void convert_wh_kernel(const float* __restrict__ Wih, const float* __restrict__ Wfh,
                       const float* __restrict__ Wuh, __bf16* __restrict__ WHb) {
  int idx = blockIdx.x * 256 + threadIdx.x;           // 3*Hv*Hv total, exact
  int gate = idx / (Hv * Hv);
  int rem  = idx - gate * (Hv * Hv);
  const float* W = (gate == 0) ? Wih : (gate == 1) ? Wfh : Wuh;
  WHb[idx] = f2bf(W[rem]);
}

// ---------------- one-time x-side GEMM -----------------------------------
// XW[g] = XEb @ W_gx^T + b_gx      [Mx, Hv] f32, g in {i,f,u}
// Wave-jobs: (Mx/32)*(Hv/32) = 256*16 = 4096 per gate; 8 waves/block.
__global__ __launch_bounds__(256)
void gemm_xw_kernel(const __bf16* __restrict__ XEb, const __bf16* __restrict__ WXb,
                    const float* __restrict__ bix, const float* __restrict__ bfx,
                    const float* __restrict__ bux, float* __restrict__ XW) {
  const int gate = blockIdx.y;
  const int lane = threadIdx.x & 31;
  const int wave = threadIdx.x >> 5;
  const int job  = blockIdx.x * 8 + wave;
  const int tn2 = job & 15;                 // Hv/32 = 16
  const int tm2 = job >> 4;                 // Mx/32 = 256
  const int m0 = tm2 * 32, n0 = tn2 * 32;

  const float* bias = (gate == 0) ? bix : (gate == 1) ? bfx : bux;
  const int lcol = lane & 15;
  float bv0 = bias[n0 + lcol];
  float bv1 = bias[n0 + 16 + lcol];
  v8f a00, a01, a10, a11;
#pragma unroll
  for (int r = 0; r < 8; ++r) { a00[r] = bv0; a10[r] = bv0; a01[r] = bv1; a11[r] = bv1; }

  wmma_block32(XEb, EPv, WXb + (size_t)gate * Hv * EPv, EPv,
               m0, n0, EPv, lane, a00, a01, a10, a11);

  float* out = XW + (size_t)gate * Mx * Hv;
  const int hi8 = (lane >> 4) << 3;
#pragma unroll
  for (int r = 0; r < 8; ++r) {
    size_t r0 = (size_t)(m0 + hi8 + r) * Hv;
    size_t r1 = (size_t)(m0 + 16 + hi8 + r) * Hv;
    out[r0 + n0 + lcol]       = a00[r];
    out[r0 + n0 + 16 + lcol]  = a01[r];
    out[r1 + n0 + lcol]       = a10[r];
    out[r1 + n0 + 16 + lcol]  = a11[r];
  }
}

// ---------------- per-step gate kernel -----------------------------------
// Uses the incrementally maintained caches HW_g[b,s,:] = all_H[b,s] @ W_gh^T.
//   si/sf/su = sum_s cmask * HW_{i,f,u};  fc = sum_s cmask*sigmoid(HWf+bfh+fx)*C
__global__ __launch_bounds__(256)
void step_kernel(const float* __restrict__ XWi, const float* __restrict__ XWf,
                 const float* __restrict__ XWu,
                 const float* __restrict__ bih, const float* __restrict__ bfh,
                 const float* __restrict__ buh,
                 const int* __restrict__ bfs, const int* __restrict__ children,
                 const float* __restrict__ HWi, const float* __restrict__ HWf,
                 const float* __restrict__ HWu, float* __restrict__ allC,
                 __bf16* __restrict__ hbf, float* __restrict__ hlast, int t) {
  const int b = blockIdx.x;
  const int n = blockIdx.y * 256 + threadIdx.x;
  const int cur = bfs[b * Sv + t];                    // uniform per block

  __shared__ float smask[Sv];
  if (threadIdx.x < Sv)
    smask[threadIdx.x] = (float)children[((size_t)b * Sv + cur) * Sv + threadIdx.x];
  __syncthreads();

  const size_t rowx = ((size_t)b * Sv + cur) * Hv + n;
  const float xi = XWi[rowx];                         // includes bix
  const float xf = XWf[rowx];                         // fx_x (includes bfx)
  const float xu = XWu[rowx];                         // includes bux
  const float bfhn = bfh[n];

  const size_t base = ((size_t)b * Sv) * Hv + n;
  const float* pHi = HWi + base;
  const float* pHf = HWf + base;
  const float* pHu = HWu + base;
  const float* pC  = allC + base;

  float si = 0.f, sf = 0.f, su = 0.f, fc = 0.f;
  for (int s = 0; s < Sv; ++s) {
    float m = smask[s];
    if (m != 0.f) {                                   // uniform branch
      size_t off = (size_t)s * Hv;
      float hi = pHi[off], hf = pHf[off], hu = pHu[off], cc = pC[off];
      si += m * hi; sf += m * hf; su += m * hu;
      fc += m * sigmoidf_(hf + bfhn + xf) * cc;       // per-child forget
    }
  }

  float ig = sigmoidf_(xi + si + bih[n]);
  float og = sigmoidf_(xf + sf + bfhn);               // 'o' reuses fx/fh weights
  float ug = tanhf(xu + su + buh[n]);
  float c  = ig * ug + fc;
  float h  = og * tanhf(c);

  allC[rowx]        = c;
  hbf[b * Hv + n]   = f2bf(h);
  hlast[b * Hv + n] = h;
}

// ---------------- per-step cache update ----------------------------------
// HW_g[b, cur[b], :] = h[b] @ W_gh^T    (M=B=128, N=K=H=512)
// Wave-jobs: (Bv/32)*(Hv/32) = 4*16 = 64 per gate; 8 waves/block.
__global__ __launch_bounds__(256)
void gemm_hw_kernel(const __bf16* __restrict__ hbf, const __bf16* __restrict__ WHb,
                    float* __restrict__ HWi, float* __restrict__ HWf,
                    float* __restrict__ HWu, const int* __restrict__ bfs, int t) {
  const int gate = blockIdx.y;
  const int lane = threadIdx.x & 31;
  const int wave = threadIdx.x >> 5;
  const int job  = blockIdx.x * 8 + wave;
  const int tn2 = job & 15;                 // Hv/32 = 16
  const int tm2 = job >> 4;                 // Bv/32 = 4
  const int m0 = tm2 * 32, n0 = tn2 * 32;

  v8f a00, a01, a10, a11;
#pragma unroll
  for (int r = 0; r < 8; ++r) { a00[r] = 0.f; a01[r] = 0.f; a10[r] = 0.f; a11[r] = 0.f; }

  wmma_block32(hbf, Hv, WHb + (size_t)gate * Hv * Hv, Hv,
               m0, n0, Hv, lane, a00, a01, a10, a11);

  float* dst = (gate == 0) ? HWi : (gate == 1) ? HWf : HWu;
  const int hi8  = (lane >> 4) << 3;
  const int lcol = lane & 15;
#pragma unroll
  for (int r = 0; r < 8; ++r) {
    int b0r = m0 + hi8 + r;
    int b1r = m0 + 16 + hi8 + r;
    int cu0 = bfs[b0r * Sv + t];
    int cu1 = bfs[b1r * Sv + t];
    size_t r0 = ((size_t)b0r * Sv + cu0) * Hv;
    size_t r1 = ((size_t)b1r * Sv + cu1) * Hv;
    dst[r0 + n0 + lcol]      = a00[r];
    dst[r0 + n0 + 16 + lcol] = a01[r];
    dst[r1 + n0 + lcol]      = a10[r];
    dst[r1 + n0 + 16 + lcol] = a11[r];
  }
}

// ---------------- final projection ---------------------------------------
__global__ __launch_bounds__(256)
void out_kernel(const float* __restrict__ hlast, const float* __restrict__ Wout,
                const float* __restrict__ bout, float* __restrict__ out) {
  __shared__ float red[256];
  const int b = blockIdx.x;
  for (int l = 0; l < Lv; ++l) {
    float acc = 0.f;
    for (int n = threadIdx.x; n < Hv; n += 256)
      acc += hlast[b * Hv + n] * Wout[l * Hv + n];
    red[threadIdx.x] = acc;
    __syncthreads();
    for (int s = 128; s > 0; s >>= 1) {
      if (threadIdx.x < s) red[threadIdx.x] += red[threadIdx.x + s];
      __syncthreads();
    }
    if (threadIdx.x == 0) out[b * Lv + l] = red[0] + bout[l];
    __syncthreads();
  }
}

// =========================================================================
extern "C" void kernel_launch(void* const* d_in, const int* in_sizes, int n_in,
                              void* d_out, int out_size, void* d_ws, size_t ws_size,
                              hipStream_t stream) {
  const int*   x        = (const int*)d_in[0];
  const int*   bfs      = (const int*)d_in[1];
  const int*   children = (const int*)d_in[2];
  const float* embed    = (const float*)d_in[3];
  const float* Wix = (const float*)d_in[4];  const float* bix = (const float*)d_in[5];
  const float* Wih = (const float*)d_in[6];  const float* bih = (const float*)d_in[7];
  const float* Wfx = (const float*)d_in[8];  const float* bfx = (const float*)d_in[9];
  const float* Wfh = (const float*)d_in[10]; const float* bfh = (const float*)d_in[11];
  const float* Wux = (const float*)d_in[12]; const float* bux = (const float*)d_in[13];
  const float* Wuh = (const float*)d_in[14]; const float* buh = (const float*)d_in[15];
  const float* Wout = (const float*)d_in[16]; const float* bout = (const float*)d_in[17];

  char* base = (char*)d_ws;
  size_t off = 0;
  auto alloc = [&](size_t bytes) -> char* {
    char* p = base + off;
    off += (bytes + 255) & ~(size_t)255;
    return p;
  };

  const size_t SZ = (size_t)Bv * Sv * Hv;             // 4,194,304 elems
  float*  allC = (float*)alloc(SZ * 4);               // contiguous zero block:
  float*  HWi  = (float*)alloc(SZ * 4);               //   allC, HWi, HWf, HWu
  float*  HWf  = (float*)alloc(SZ * 4);
  float*  HWu  = (float*)alloc(SZ * 4);
  float*  XW   = (float*)alloc(3 * SZ * 4);           // [3][Mx][Hv]
  __bf16* XEb  = (__bf16*)alloc((size_t)Mx * EPv * 2);
  __bf16* WXb  = (__bf16*)alloc((size_t)3 * Hv * EPv * 2);
  __bf16* WHb  = (__bf16*)alloc((size_t)3 * Hv * Hv * 2);
  __bf16* hbf  = (__bf16*)alloc((size_t)Bv * Hv * 2);
  float*  hlast = (float*)alloc((size_t)Bv * Hv * 4);
  (void)ws_size; (void)n_in; (void)in_sizes; (void)out_size;

  // Zero all_C + the three HW caches (one contiguous 64 MB memset).
  hipMemsetAsync(allC, 0, 4 * SZ * 4, stream);

  // Stage bf16 operands.
  embed_gather_kernel<<<(Mx * EPv) / 256, 256, 0, stream>>>(embed, x, XEb);
  convert_wx_kernel<<<(3 * Hv * EPv) / 256, 256, 0, stream>>>(Wix, Wfx, Wux, WXb);
  convert_wh_kernel<<<(3 * Hv * Hv) / 256, 256, 0, stream>>>(Wih, Wfh, Wuh, WHb);

  // One-time x-side projections: XW[g] = XE @ W_gx^T + b_gx.
  gemm_xw_kernel<<<dim3((Mx / 32) * (Hv / 32) / 8, 3), 256, 0, stream>>>(
      XEb, WXb, bix, bfx, bux, XW);

  // Sequential BFS steps.
  const float* XWi = XW;
  const float* XWf = XW + SZ;
  const float* XWu = XW + 2 * SZ;
  for (int t = 0; t < Sv; ++t) {
    step_kernel<<<dim3(Bv, Hv / 256), 256, 0, stream>>>(
        XWi, XWf, XWu, bih, bfh, buh, bfs, children,
        HWi, HWf, HWu, allC, hbf, hlast, t);
    gemm_hw_kernel<<<dim3((Bv / 32) * (Hv / 32) / 8, 3), 256, 0, stream>>>(
        hbf, WHb, HWi, HWf, HWu, bfs, t);
  }

  out_kernel<<<Bv, 256, 0, stream>>>(hlast, Wout, bout, (float*)d_out);
}
